// DetectModel_60584808677382
// MI455X (gfx1250) — compile-verified, matrix operand order checked by reading, not run on hardware
//
#include <hip/hip_runtime.h>
#include <hip/hip_bf16.h>

// ---------------------------------------------------------------------------
// VoteNet-style detector for MI455X (gfx1250, wave32, WMMA).
// Dense math runs through v_wmma_f32_16x16x32_f16 with fragment-swizzled,
// zero-padded f16 weights (packed once per call) so the inner loops contain
// only unconditional b128 loads + WMMA.
// ---------------------------------------------------------------------------

typedef __attribute__((ext_vector_type(16))) _Float16 v16h;
typedef __attribute__((ext_vector_type(8)))  _Float16 v8h;
typedef __attribute__((ext_vector_type(8)))  float    v8f;

#define BB 4
#define NN 20000
#define NUM_PROPOSAL 256
#define LOGIT_C 119   // 2+3+12+12+18+54+18
#define OUT_C 188     // 2+3+3+12+12+12+18+54+54+18

__host__ __device__ static inline int pad64(int x) { return (x + 63) & ~63; }

// A-fragment for v_wmma_f32_16x16x32_f16: lane's row, two contiguous 8xf16
// chunks at f16 offsets k0+half*8 and k0+16+half*8 (ISA 7.12.2 layout).
__device__ __forceinline__ v16h make_afrag(const _Float16* rowp, int k0, int half) {
  v8h lo = *(const v8h*)(rowp + k0 + half * 8);
  v8h hi = *(const v8h*)(rowp + k0 + 16 + half * 8);
  return __builtin_shufflevector(lo, hi, 0,1,2,3,4,5,6,7,8,9,10,11,12,13,14,15);
}

// ========================= Weight packing ===================================
// dst is fragment-swizzled: dst[((ks*tilesN + tn)*512) + lane*16 + e]
// holds W[kk, col] (f16, zero padded), kk/col per the WMMA B layout.
__global__ void pack_w_kernel(const float* __restrict__ W, int K, int N,
                              _Float16* __restrict__ dst, int tilesN, int total) {
  int t = blockIdx.x * blockDim.x + threadIdx.x;
  if (t >= total) return;
  int le = t & 511;
  int lane = le >> 4, e = le & 15;
  int tile = t >> 9;
  int ks = tile / tilesN, tn = tile - ks * tilesN;
  int vr = e >> 1;
  int kk = ks * 32 + ((vr < 4) ? 0 : 16) + (lane >> 4) * 8 + ((vr & 3) << 1) + (e & 1);
  int col = tn * 16 + (lane & 15);
  float v = (kk < K && col < N) ? W[(size_t)kk * N + col] : 0.0f;
  dst[t] = (_Float16)v;
}

__global__ void pack_vec_kernel(const float* __restrict__ src, int n,
                                float* __restrict__ dst, int np) {
  int t = blockIdx.x * blockDim.x + threadIdx.x;
  if (t >= np) return;
  dst[t] = (t < n) ? src[t] : 0.0f;
}

// ========================= FPS ==============================================
__global__ __launch_bounds__(256)
void fps_kernel(const float* __restrict__ pts, int Np, int strideP,
                int nOut, int* __restrict__ outIdx, float* __restrict__ distWS) {
  int b = blockIdx.x;
  const float* P = pts + (size_t)b * strideP * 3;
  float* dist = distWS + (size_t)b * Np;
  __shared__ float rv[256];
  __shared__ int   ri[256];
  __shared__ int   selS;
  int tid = threadIdx.x;
  for (int i = tid; i < Np; i += blockDim.x) dist[i] = 1e10f;
  __syncthreads();
  int sel = 0;
  for (int t = 0; t < nOut; ++t) {
    if (tid == 0) outIdx[b * nOut + t] = sel;
    float lx = P[sel*3+0], ly = P[sel*3+1], lz = P[sel*3+2];
    float bm = -1.0f; int bi = 0;
    for (int i = tid; i < Np; i += blockDim.x) {
      float dx = P[i*3+0]-lx, dy = P[i*3+1]-ly, dz = P[i*3+2]-lz;
      float d = fminf(dist[i], dx*dx + dy*dy + dz*dz);
      dist[i] = d;
      if (d > bm) { bm = d; bi = i; }
    }
    rv[tid] = bm; ri[tid] = bi;
    __syncthreads();
    for (int s = blockDim.x >> 1; s > 0; s >>= 1) {
      if (tid < s) {
        if (rv[tid+s] > rv[tid] || (rv[tid+s] == rv[tid] && ri[tid+s] < ri[tid])) {
          rv[tid] = rv[tid+s]; ri[tid] = ri[tid+s];
        }
      }
      __syncthreads();
    }
    if (tid == 0) selS = ri[0];
    __syncthreads();
    sel = selS;
  }
}

// ========================= Ball query =======================================
__global__ __launch_bounds__(32)
void ball_kernel(const float* __restrict__ pts, int Np, int strideP,
                 const float* __restrict__ refs, int strideR, int Nr,
                 int k, float r2, int* __restrict__ idxOut) {
  int g = blockIdx.x;
  int b = g / Nr, r = g % Nr;
  const float* P = pts + (size_t)b * strideP * 3;
  const float* R = refs + ((size_t)b * strideR + r) * 3;
  __shared__ int flags[32];
  __shared__ int found[64];
  __shared__ int cntS;
  int lane = threadIdx.x;
  if (lane == 0) cntS = 0;
  __syncthreads();
  float rx = R[0], ry = R[1], rz = R[2];
  for (int base = 0; base < Np; base += 32) {
    int i = base + lane;
    int ok = 0;
    if (i < Np) {
      float dx = P[i*3+0]-rx, dy = P[i*3+1]-ry, dz = P[i*3+2]-rz;
      ok = (dx*dx + dy*dy + dz*dz) <= r2;
    }
    flags[lane] = ok;
    __syncthreads();
    if (lane == 0) {
      int c = cntS;
      for (int j = 0; j < 32 && c < k; ++j)
        if (flags[j]) found[c++] = base + j;
      cntS = c;
    }
    __syncthreads();
    if (cntS >= k) break;
  }
  __syncthreads();
  int cnt = cntS;
  int f0 = (cnt > 0) ? found[0] : 0;
  for (int c = lane; c < k; c += 32)
    idxOut[(size_t)g * k + c] = (c < cnt) ? found[c] : f0;
}

// ========================= Simple gather (xyz) ==============================
__global__ void gather_pts_kernel(const float* __restrict__ src, int strideS,
                                  const int* __restrict__ idx, int Nr,
                                  float* __restrict__ dst, int total) {
  int g = blockIdx.x * blockDim.x + threadIdx.x;
  if (g >= total) return;
  int b = g / Nr, r = g % Nr;
  int id = idx[b * Nr + r];
  const float* s = src + ((size_t)b * strideS + id) * 3;
  float* d = dst + (size_t)g * 3;
  d[0] = s[0]; d[1] = s[1]; d[2] = s[2];
}

// ========================= Fused gather+MLP+maxpool (WMMA) ==================
// One block (4 waves) per ref point. Activations live in LDS as f16 with
// pad64 widths; weights come pre-swizzled from global (unconditional loads).
__global__ __launch_bounds__(128)
void aggregate_wmma(const float* __restrict__ pts, int strideP,
                    const float* __restrict__ feats, int Cf, int strideF,
                    const float* __restrict__ refs, int strideR, int Nr,
                    const int* __restrict__ ballIdx, int k, float invRadius,
                    const _Float16* __restrict__ W0, const float* __restrict__ s0,
                    const float* __restrict__ b0, int c0,
                    const _Float16* __restrict__ W1, const float* __restrict__ s1,
                    const float* __restrict__ b1, int c1,
                    const _Float16* __restrict__ W2, const float* __restrict__ s2,
                    const float* __restrict__ b2, int c2,
                    float* __restrict__ out) {
  extern __shared__ float smemf[];
  _Float16* smem = (_Float16*)smemf;
  int g = blockIdx.x;
  int b = g / Nr, r = g % Nr;
  int cin = 3 + Cf;
  int cinP = pad64(cin);
  int c0P = pad64(c0), c1P = pad64(c1), c2P = pad64(c2);
  int rowsP = (k + 15) & ~15;
  int maxW = cinP; if (c0P > maxW) maxW = c0P; if (c1P > maxW) maxW = c1P; if (c2P > maxW) maxW = c2P;
  _Float16* buf0 = smem;
  _Float16* buf1 = smem + (size_t)rowsP * maxW;

  const int* idx = ballIdx + (size_t)g * k;
  const float* R = refs + ((size_t)b * strideR + r) * 3;
  float rx = R[0], ry = R[1], rz = R[2];
  const float* P = pts + (size_t)b * strideP * 3;
  const float* F = feats + (size_t)b * strideF * Cf;  // dummy but valid when Cf==0
  int tid = threadIdx.x;

  // Stage X (rowsP x cinP, zero padded) into buf0 as f16. One-time cost.
  for (int t = tid; t < rowsP * cinP; t += blockDim.x) {
    int row = t / cinP, c = t - row * cinP;
    float v = 0.0f;
    if (row < k && c < cin) {
      int id = idx[row];
      if (c < 3) {
        float rc = (c == 0) ? rx : (c == 1) ? ry : rz;
        v = (P[id*3 + c] - rc) * invRadius;
      } else {
        v = F[(size_t)id * Cf + (c - 3)];
      }
    }
    buf0[row * cinP + c] = (_Float16)v;
  }
  __syncthreads();

  const _Float16* Ws[3] = {W0, W1, W2};
  const float* Ss[3] = {s0, s1, s2};
  const float* Bs[3] = {b0, b1, b2};
  int KPL[3] = {cinP, c0P, c1P};
  int NPL[3] = {c0P, c1P, c2P};

  _Float16* cur = buf0;
  _Float16* nxt = buf1;
  int lane = tid & 31, wv = tid >> 5, nw = blockDim.x >> 5;
  int half = lane >> 4, l15 = lane & 15;

  for (int l = 0; l < 3; ++l) {
    int Kp = KPL[l], NcP = NPL[l];
    const _Float16* W = Ws[l];
    const float* S = Ss[l]; const float* Bv = Bs[l];
    int tilesN = NcP >> 4;
    int T = (rowsP >> 4) * tilesN;
    for (int t = wv; t < T; t += nw) {
      int tm = t / tilesN, tn = t - tm * tilesN;
      const _Float16* arowp = cur + (size_t)(tm * 16 + l15) * Kp;
      v8f acc = {};
      for (int k0 = 0; k0 < Kp; k0 += 32) {
        v16h afrag = make_afrag(arowp, k0, half);
        v16h bfrag = *(const v16h*)(W + ((size_t)((k0 >> 5) * tilesN + tn)) * 512 + lane * 16);
        acc = __builtin_amdgcn_wmma_f32_16x16x32_f16(false, afrag, false, bfrag,
                                                     (short)0, acc, false, false);
      }
      int col = tn * 16 + l15;
      float sc = S[col], bi = Bv[col];   // padded: zeros beyond Nc
#pragma unroll
      for (int rr = 0; rr < 8; ++rr) {
        int row = tm * 16 + half * 8 + rr;
        nxt[(size_t)row * NcP + col] = (_Float16)fmaxf(acc[rr] * sc + bi, 0.0f);
      }
    }
    __syncthreads();
    _Float16* tb = cur; cur = nxt; nxt = tb;
  }

  // Max-pool over the k valid neighbor rows.
  for (int c = tid; c < c2; c += blockDim.x) {
    float m = -3.0e38f;
    for (int row = 0; row < k; ++row)
      m = fmaxf(m, (float)cur[(size_t)row * c2P + c]);
    out[(size_t)g * c2 + c] = m;
  }
}

// ========================= Generic WMMA GEMM + epilogue =====================
// 128 threads = 4 waves compute a 64x64 output tile (4 WMMA accs per wave).
// A: MxK f32 tight row-major; W pre-swizzled padded f16 (Kp x NP).
// mode 1: relu(A*W*scale + bias) ; mode 2: A*W + bias
__global__ __launch_bounds__(128)
void gemm_wmma(const float* __restrict__ A, int M, int K,
               const _Float16* __restrict__ Wsw, int tilesN,
               const float* __restrict__ scaleP, const float* __restrict__ biasP,
               float* __restrict__ Y, int Ncols, int mode) {
  __shared__ _Float16 As[64 * 32];
  int bn = blockIdx.x, bm = blockIdx.y;
  int tid = threadIdx.x, lane = tid & 31, wv = tid >> 5;
  int half = lane >> 4, l15 = lane & 15;
  int rowBase = bm * 64;
  int Kp = pad64(K);
  v8f z = {};
  v8f a0 = z, a1 = z, a2 = z, a3 = z;
  for (int k0 = 0; k0 < Kp; k0 += 32) {
    __syncthreads();
    if (rowBase + 64 <= M && k0 + 32 <= K) {
      // fast path: every element in range (all pipeline M%64==0, K%32==0)
      int row = tid >> 1, ch = (tid & 1) * 16;
      const float4* src = (const float4*)(A + (size_t)(rowBase + row) * K + k0 + ch);
      _Float16* d = As + row * 32 + ch;
#pragma unroll
      for (int q = 0; q < 4; ++q) {
        float4 f = src[q];
        d[q*4+0] = (_Float16)f.x; d[q*4+1] = (_Float16)f.y;
        d[q*4+2] = (_Float16)f.z; d[q*4+3] = (_Float16)f.w;
      }
    } else {
      for (int t = tid; t < 64 * 32; t += 128) {
        int rr = t >> 5, cc = t & 31;
        int grow = rowBase + rr, gcol = k0 + cc;
        float v = (grow < M && gcol < K) ? A[(size_t)grow * K + gcol] : 0.0f;
        As[rr * 32 + cc] = (_Float16)v;
      }
    }
    __syncthreads();
    v16h afrag = make_afrag(As + (size_t)(wv * 16 + l15) * 32, 0, half);
    const _Float16* wbase = Wsw + ((size_t)(k0 >> 5) * tilesN + bn * 4) * 512 + lane * 16;
    v16h bf0 = *(const v16h*)(wbase);
    v16h bf1 = *(const v16h*)(wbase + 512);
    v16h bf2 = *(const v16h*)(wbase + 1024);
    v16h bf3 = *(const v16h*)(wbase + 1536);
    a0 = __builtin_amdgcn_wmma_f32_16x16x32_f16(false, afrag, false, bf0, (short)0, a0, false, false);
    a1 = __builtin_amdgcn_wmma_f32_16x16x32_f16(false, afrag, false, bf1, (short)0, a1, false, false);
    a2 = __builtin_amdgcn_wmma_f32_16x16x32_f16(false, afrag, false, bf2, (short)0, a2, false, false);
    a3 = __builtin_amdgcn_wmma_f32_16x16x32_f16(false, afrag, false, bf3, (short)0, a3, false, false);
  }
  int row0 = rowBase + wv * 16 + half * 8;
  v8f accs[4] = {a0, a1, a2, a3};
#pragma unroll
  for (int t = 0; t < 4; ++t) {
    int col = (bn * 4 + t) * 16 + l15;
    if (col >= Ncols) continue;
    float sc = (mode == 1) ? scaleP[col] : 1.0f;
    float bi = biasP[col];
#pragma unroll
    for (int rr = 0; rr < 8; ++rr) {
      int row = row0 + rr;
      if (row < M) {
        float v = accs[t][rr] * sc + bi;
        if (mode == 1) v = fmaxf(v, 0.0f);
        Y[(size_t)row * Ncols + col] = v;
      }
    }
  }
}

// ========================= 3-NN interpolate + concat ========================
__global__ void interp_concat_kernel(const float* __restrict__ unk, int strideU, int Nu,
                                     const float* __restrict__ kn, int strideK, int Nk,
                                     const float* __restrict__ knF,
                                     const float* __restrict__ ownF, int C,
                                     float* __restrict__ X, int total) {
  int g = blockIdx.x * blockDim.x + threadIdx.x;
  if (g >= total) return;
  int b = g / Nu, u = g % Nu;
  const float* U = unk + ((size_t)b * strideU + u) * 3;
  float ux = U[0], uy = U[1], uz = U[2];
  const float* Kp = kn + (size_t)b * strideK * 3;
  float d0 = 3e38f, d1 = 3e38f, d2 = 3e38f;
  int i0 = 0, i1 = 0, i2 = 0;
  for (int i = 0; i < Nk; ++i) {
    float dx = Kp[i*3+0]-ux, dy = Kp[i*3+1]-uy, dz = Kp[i*3+2]-uz;
    float d = fmaxf(dx*dx + dy*dy + dz*dz, 0.0f);
    if (d < d0)      { d2 = d1; i2 = i1; d1 = d0; i1 = i0; d0 = d; i0 = i; }
    else if (d < d1) { d2 = d1; i2 = i1; d1 = d;  i1 = i; }
    else if (d < d2) { d2 = d;  i2 = i; }
  }
  d0 = fmaxf(d0, 1e-10f); d1 = fmaxf(d1, 1e-10f); d2 = fmaxf(d2, 1e-10f);
  float w0 = 1.0f/d0, w1 = 1.0f/d1, w2 = 1.0f/d2;
  float wsum = w0 + w1 + w2;
  w0 /= wsum; w1 /= wsum; w2 /= wsum;
  const float* F0 = knF + ((size_t)b * Nk + i0) * C;
  const float* F1 = knF + ((size_t)b * Nk + i1) * C;
  const float* F2 = knF + ((size_t)b * Nk + i2) * C;
  const float* own = ownF + (size_t)g * C;
  float* row = X + (size_t)g * (2 * C);
  for (int c = 0; c < C; ++c) {
    row[c]     = own[c];
    row[C + c] = w0 * F0[c] + w1 * F1[c] + w2 * F2[c];
  }
}

// ========================= Vote head ========================================
__global__ void vote_kernel(const float* __restrict__ s1pts,
                            const float* __restrict__ s2fts,
                            const float* __restrict__ net,
                            float* __restrict__ vxyz, float* __restrict__ vfts,
                            int total) {
  int g = blockIdx.x * blockDim.x + threadIdx.x;
  if (g >= total) return;
  int b = g / 1024, u = g - b * 1024;
  const float* pp = s1pts + ((size_t)b * 2048 + u) * 3;
  const float* nn = net + (size_t)g * 259;
  vxyz[g*3+0] = pp[0] + nn[0];
  vxyz[g*3+1] = pp[1] + nn[1];
  vxyz[g*3+2] = pp[2] + nn[2];
  for (int c = 0; c < 256; ++c)
    vfts[(size_t)g * 256 + c] = s2fts[(size_t)g * 256 + c] + nn[3 + c];
}

// ========================= Output assembly ==================================
__global__ void assemble_kernel(const float* __restrict__ logits,
                                const float* __restrict__ aggP,
                                const float* __restrict__ meanSize,
                                float* __restrict__ out, int total) {
  int g = blockIdx.x * blockDim.x + threadIdx.x;
  if (g >= total) return;
  const float* L = logits + (size_t)g * LOGIT_C;
  const float* A = aggP + (size_t)g * 3;
  float* O = out + (size_t)g * OUT_C;
  const float hscale = 3.14159265358979323846f / 12.0f;
  O[0] = L[0]; O[1] = L[1];
  O[2] = A[0]; O[3] = A[1]; O[4] = A[2];
  O[5] = A[0] + L[2]; O[6] = A[1] + L[3]; O[7] = A[2] + L[4];
  for (int i = 0; i < 12; ++i) {
    O[8  + i] = L[5  + i];
    O[20 + i] = L[17 + i];
    O[32 + i] = L[17 + i] * hscale;
  }
  for (int i = 0; i < 18; ++i) O[44 + i] = L[29 + i];
  for (int i = 0; i < 54; ++i) {
    O[62  + i] = L[47 + i];
    O[116 + i] = L[47 + i] * meanSize[i];
  }
  for (int i = 0; i < 18; ++i) O[170 + i] = L[101 + i];
}

// ===========================================================================
extern "C" void kernel_launch(void* const* d_in, const int* in_sizes, int n_in,
                              void* d_out, int out_size, void* d_ws, size_t ws_size,
                              hipStream_t stream) {
  (void)in_sizes; (void)n_in; (void)out_size; (void)ws_size;
  auto F = [&](int i) { return (const float*)d_in[i]; };

  const float* points   = F(0);   // (4,20000,3)
  const float* meanSize = F(1);   // (18,3)
  struct Dense { const float* W; const float* s; const float* b; };
  auto dn = [&](int base) { return Dense{F(base), F(base+1), F(base+2)}; };
  Dense ds1[3] = {dn(2),  dn(5),  dn(8)};
  Dense ds2[3] = {dn(11), dn(14), dn(17)};
  Dense ds3[3] = {dn(20), dn(23), dn(26)};
  Dense ds4[3] = {dn(29), dn(32), dn(35)};
  Dense us1[3] = {dn(38), dn(41), dn(44)};
  Dense us2[3] = {dn(47), dn(50), dn(53)};
  Dense vmlp[3]= {dn(56), dn(59), dn(62)};
  const float* voW = F(65); const float* voB = F(66);
  Dense pagg[3] = {dn(67), dn(70), dn(73)};
  Dense pmlp[3] = {dn(76), dn(79), dn(82)};
  const float* poW = F(85); const float* poB = F(86);

  // ---- workspace carve-out -------------------------------------------------
  char* wsb = (char*)d_ws;
  size_t off = 0;
  auto alloc = [&](size_t bytes) -> void* {
    void* p = wsb + off;
    off = (off + bytes + 255) & ~(size_t)255;
    return p;
  };
  float* dist     = (float*)alloc((size_t)BB * NN * 4);
  int*   s1_idx   = (int*)  alloc((size_t)BB * 2048 * 4);
  float* s1_pts   = (float*)alloc((size_t)BB * 2048 * 3 * 4);
  int*   ball1    = (int*)  alloc((size_t)BB * 2048 * 64 * 4);
  float* s1_fts   = (float*)alloc((size_t)BB * 2048 * 128 * 4);
  int*   ball2    = (int*)  alloc((size_t)BB * 1024 * 32 * 4);
  float* s2_fts   = (float*)alloc((size_t)BB * 1024 * 256 * 4);
  int*   ball3    = (int*)  alloc((size_t)BB * 512 * 16 * 4);
  float* s3_fts   = (float*)alloc((size_t)BB * 512 * 256 * 4);
  int*   ball4    = (int*)  alloc((size_t)BB * 256 * 16 * 4);
  float* s4_fts   = (float*)alloc((size_t)BB * 256 * 256 * 4);
  float* usin     = (float*)alloc((size_t)BB * 1024 * 512 * 4);
  float* hbuf0    = (float*)alloc((size_t)4096 * 256 * 4);
  float* hbuf1    = (float*)alloc((size_t)4096 * 256 * 4);
  float* vote_net = (float*)alloc((size_t)BB * 1024 * 259 * 4);
  float* vote_xyz = (float*)alloc((size_t)BB * 1024 * 3 * 4);
  float* vote_fts = (float*)alloc((size_t)BB * 1024 * 256 * 4);
  int*   ref_idx  = (int*)  alloc((size_t)BB * 256 * 4);
  float* agg_pts  = (float*)alloc((size_t)BB * 256 * 3 * 4);
  int*   ball5    = (int*)  alloc((size_t)BB * 256 * 16 * 4);
  float* agg_fts  = (float*)alloc((size_t)BB * 256 * 128 * 4);
  float* logits   = (float*)alloc((size_t)BB * 256 * LOGIT_C * 4);

  // ---- pack weights (fragment-swizzled f16) + padded scale/bias -----------
  struct Packed { const _Float16* W; const float* s; const float* b;
                  int N; int NP; int tilesN; };
  auto packDense = [&](Dense d, int K, int N, bool bn) -> Packed {
    Packed p;
    int Kp = pad64(K), NP = pad64(N);
    p.N = N; p.NP = NP; p.tilesN = NP / 16;
    size_t welems = (size_t)Kp * NP;
    _Float16* w = (_Float16*)alloc(welems * 2);
    pack_w_kernel<<<((int)welems + 255) / 256, 256, 0, stream>>>(
        d.W, K, N, w, p.tilesN, (int)welems);
    p.W = w;
    float* sv = nullptr;
    if (bn) {
      sv = (float*)alloc((size_t)NP * 4);
      pack_vec_kernel<<<(NP + 255) / 256, 256, 0, stream>>>(d.s, N, sv, NP);
    }
    float* bv = (float*)alloc((size_t)NP * 4);
    pack_vec_kernel<<<(NP + 255) / 256, 256, 0, stream>>>(d.b, N, bv, NP);
    p.s = sv; p.b = bv;
    return p;
  };

  Packed Pds1[3] = {packDense(ds1[0],3,64,true),    packDense(ds1[1],64,64,true),   packDense(ds1[2],64,128,true)};
  Packed Pds2[3] = {packDense(ds2[0],131,128,true), packDense(ds2[1],128,128,true), packDense(ds2[2],128,256,true)};
  Packed Pds3[3] = {packDense(ds3[0],259,128,true), packDense(ds3[1],128,128,true), packDense(ds3[2],128,256,true)};
  Packed Pds4[3] = {packDense(ds4[0],259,128,true), packDense(ds4[1],128,128,true), packDense(ds4[2],128,256,true)};
  Packed Pus1[3] = {packDense(us1[0],512,256,true), packDense(us1[1],256,128,true), packDense(us1[2],128,256,true)};
  Packed Pus2[3] = {packDense(us2[0],512,256,true), packDense(us2[1],256,128,true), packDense(us2[2],128,256,true)};
  Packed Pvm[3]  = {packDense(vmlp[0],256,256,true),packDense(vmlp[1],256,256,true),packDense(vmlp[2],256,256,true)};
  Packed Pvo     = packDense(Dense{voW, nullptr, voB}, 256, 259, false);
  Packed Ppa[3]  = {packDense(pagg[0],259,128,true), packDense(pagg[1],128,128,true), packDense(pagg[2],128,128,true)};
  Packed Ppm[3]  = {packDense(pmlp[0],128,128,true), packDense(pmlp[1],128,128,true), packDense(pmlp[2],128,128,true)};
  Packed Ppo     = packDense(Dense{poW, nullptr, poB}, 128, LOGIT_C, false);

  auto gemm = [&](const float* A, int M, int K, const Packed& P, float* Y, int mode) {
    dim3 grid(P.NP / 64, (M + 63) / 64);
    gemm_wmma<<<grid, 128, 0, stream>>>(A, M, K, P.W, P.tilesN, P.s, P.b, Y, P.N, mode);
  };
  auto mlp3 = [&](const float* X, int M, int K, Packed* L,
                  int c0, int c1, float* dst) {
    gemm(X,     M, K,  L[0], hbuf0, 1);
    gemm(hbuf0, M, c0, L[1], hbuf1, 1);
    gemm(hbuf1, M, c1, L[2], dst,   1);
  };
  auto aggregate = [&](const float* pts, int strideP,
                       const float* feats, int Cf, int strideF,
                       const float* refs, int strideR, int Nr,
                       const int* bidx, int k, float radius,
                       Packed* L, int c0, int c1, int c2, float* outp) {
    int cinP = pad64(3 + Cf);
    int c0P = pad64(c0), c1P = pad64(c1), c2P = pad64(c2);
    int maxW = cinP; if (c0P > maxW) maxW = c0P; if (c1P > maxW) maxW = c1P; if (c2P > maxW) maxW = c2P;
    int rowsP = (k + 15) & ~15;
    size_t shmem = (size_t)2 * rowsP * maxW * sizeof(_Float16);
    aggregate_wmma<<<BB * Nr, 128, shmem, stream>>>(
        pts, strideP, feats, Cf, strideF, refs, strideR, Nr, bidx, k,
        1.0f / radius,
        L[0].W, L[0].s, L[0].b, c0,
        L[1].W, L[1].s, L[1].b, c1,
        L[2].W, L[2].s, L[2].b, c2, outp);
  };

  // ---- stage 1: FPS 20000 -> 2048, gather, ball, aggregate ----------------
  fps_kernel<<<BB, 256, 0, stream>>>(points, NN, NN, 2048, s1_idx, dist);
  gather_pts_kernel<<<(BB*2048 + 255)/256, 256, 0, stream>>>(
      points, NN, s1_idx, 2048, s1_pts, BB*2048);
  ball_kernel<<<BB*2048, 32, 0, stream>>>(points, NN, NN, s1_pts, 2048, 2048,
                                          64, 0.2f*0.2f, ball1);
  aggregate(points, NN, points /*dummy*/, 0, 0, s1_pts, 2048, 2048, ball1, 64, 0.2f,
            Pds1, 64, 64, 128, s1_fts);

  // ---- stage 2 ------------------------------------------------------------
  ball_kernel<<<BB*1024, 32, 0, stream>>>(s1_pts, 2048, 2048, s1_pts, 2048, 1024,
                                          32, 0.4f*0.4f, ball2);
  aggregate(s1_pts, 2048, s1_fts, 128, 2048, s1_pts, 2048, 1024, ball2, 32, 0.4f,
            Pds2, 128, 128, 256, s2_fts);

  // ---- stage 3 ------------------------------------------------------------
  ball_kernel<<<BB*512, 32, 0, stream>>>(s1_pts, 1024, 2048, s1_pts, 2048, 512,
                                         16, 0.8f*0.8f, ball3);
  aggregate(s1_pts, 2048, s2_fts, 256, 1024, s1_pts, 2048, 512, ball3, 16, 0.8f,
            Pds3, 128, 128, 256, s3_fts);

  // ---- stage 4 ------------------------------------------------------------
  ball_kernel<<<BB*256, 32, 0, stream>>>(s1_pts, 512, 2048, s1_pts, 2048, 256,
                                         16, 1.2f*1.2f, ball4);
  aggregate(s1_pts, 2048, s3_fts, 256, 512, s1_pts, 2048, 256, ball4, 16, 1.2f,
            Pds4, 128, 128, 256, s4_fts);

  // ---- feature propagation us1: s4 -> s3 ----------------------------------
  interp_concat_kernel<<<(BB*512 + 255)/256, 256, 0, stream>>>(
      s1_pts, 2048, 512, s1_pts, 2048, 256, s4_fts, s3_fts, 256, usin, BB*512);
  mlp3(usin, BB*512, 512, Pus1, 256, 128, s3_fts);

  // ---- feature propagation us2: s3 -> s2 ----------------------------------
  interp_concat_kernel<<<(BB*1024 + 255)/256, 256, 0, stream>>>(
      s1_pts, 2048, 1024, s1_pts, 2048, 512, s3_fts, s2_fts, 256, usin, BB*1024);
  mlp3(usin, BB*1024, 512, Pus2, 256, 128, s2_fts);

  // ---- vote head ----------------------------------------------------------
  gemm(s2_fts, BB*1024, 256, Pvm[0], hbuf0, 1);
  gemm(hbuf0,  BB*1024, 256, Pvm[1], hbuf1, 1);
  gemm(hbuf1,  BB*1024, 256, Pvm[2], hbuf0, 1);
  gemm(hbuf0,  BB*1024, 256, Pvo, vote_net, 2);
  vote_kernel<<<(BB*1024 + 255)/256, 256, 0, stream>>>(
      s1_pts, s2_fts, vote_net, vote_xyz, vote_fts, BB*1024);

  // ---- proposal sampling + aggregation ------------------------------------
  fps_kernel<<<BB, 256, 0, stream>>>(vote_xyz, 1024, 1024, NUM_PROPOSAL, ref_idx, dist);
  gather_pts_kernel<<<(BB*256 + 255)/256, 256, 0, stream>>>(
      vote_xyz, 1024, ref_idx, 256, agg_pts, BB*256);
  ball_kernel<<<BB*256, 32, 0, stream>>>(vote_xyz, 1024, 1024, agg_pts, 256, 256,
                                         16, 0.3f*0.3f, ball5);
  aggregate(vote_xyz, 1024, vote_fts, 256, 1024, agg_pts, 256, 256, ball5, 16, 0.3f,
            Ppa, 128, 128, 128, agg_fts);

  // ---- proposal head ------------------------------------------------------
  gemm(agg_fts, BB*256, 128, Ppm[0], hbuf0, 1);
  gemm(hbuf0,   BB*256, 128, Ppm[1], hbuf1, 1);
  gemm(hbuf1,   BB*256, 128, Ppm[2], hbuf0, 1);
  gemm(hbuf0,   BB*256, 128, Ppo, logits, 2);

  assemble_kernel<<<(BB*256 + 255)/256, 256, 0, stream>>>(
      logits, agg_pts, meanSize, (float*)d_out, BB*256);
}
